// _GRU_6768868458932
// MI455X (gfx1250) — matrix operand order repaired and checked by reading, not compiled, not run on hardware
//
#include <hip/hip_runtime.h>
#include <hip/hip_bf16.h>

// ---------------- constants (match reference) ----------------
#define T_STEPS 2048
#define CIN     64
#define HID     256
#define G3      768    // 3*H

typedef __attribute__((ext_vector_type(16))) __bf16        v16bf;
typedef __attribute__((ext_vector_type(8)))  float         v8f;
typedef __attribute__((ext_vector_type(4)))  unsigned int  v4u;
typedef __attribute__((ext_vector_type(4)))  int           v4i;
typedef __attribute__((ext_vector_type(8)))  int           v8i;

union Frag {
    v16bf v;
    uint4 q[2];
};

#if defined(__gfx1250__) && __has_builtin(__builtin_amdgcn_tensor_load_to_lds) && \
    __has_builtin(__builtin_amdgcn_s_wait_tensorcnt)
#define USE_TDM 1
#else
#define USE_TDM 0
#endif

__device__ inline unsigned short f2bf(float f) {
    unsigned int u = __builtin_bit_cast(unsigned int, f);
    unsigned int r = (u + 0x7FFFu + ((u >> 16) & 1u)) >> 16;   // RNE
    return (unsigned short)r;
}
__device__ inline float sigmoidf_(float x) { return 1.0f / (1.0f + __expf(-x)); }
__device__ inline float tanhf_(float x) {
    x = fminf(fmaxf(x, -15.0f), 15.0f);
    float e = __expf(2.0f * x);
    return (e - 1.0f) / (e + 1.0f);
}

#if USE_TDM
// Issue one TDM 2D tile load: `rows` rows of `cols` f32, global row stride
// `rowStride` elements, into LDS byte offset lds_off (rows packed contiguously).
// D# per CDNA5 ISA 8.3/8.4: group0 = {count/type/addr}, group1 = {dims/strides}.
__device__ __attribute__((always_inline)) inline void tdm_load_2d_f32(
    const void* gaddr, unsigned lds_off, unsigned rows, unsigned cols, unsigned rowStride) {
    unsigned long long ga = (unsigned long long)gaddr;
    v4u g0;
    g0[0] = 1u;                                              // count=1, is_restore=0
    g0[1] = lds_off;                                         // lds_addr (bytes)
    g0[2] = (unsigned)(ga & 0xFFFFFFFFu);                    // global_addr lo
    g0[3] = (unsigned)((ga >> 32) & 0x01FFFFFFu) | 0x80000000u;  // addr hi | type=2
    v8i g1;
    g1[0] = (int)(2u << 16);                                 // data_size = 4B
    g1[1] = (int)((cols & 0xFFFFu) << 16);                   // tensor_dim0 lo16
    g1[2] = (int)(((cols >> 16) & 0xFFFFu) | ((rows & 0xFFFFu) << 16)); // dim0 hi | dim1 lo
    g1[3] = (int)(((rows >> 16) & 0xFFFFu) | ((cols & 0xFFFFu) << 16)); // dim1 hi | tile_dim0
    g1[4] = (int)(rows & 0xFFFFu);                           // tile_dim1 (tile_dim2=0)
    g1[5] = (int)rowStride;                                  // tensor_dim0_stride lo32
    g1[6] = 0;                                               // stride hi | dim1_stride lo
    g1[7] = 0;
    v4i z4 = {0, 0, 0, 0};
#if __clang_major__ >= 23
    v8i z8 = {0, 0, 0, 0, 0, 0, 0, 0};
    __builtin_amdgcn_tensor_load_to_lds(g0, g1, z4, z4, z8, 0);
#else
    __builtin_amdgcn_tensor_load_to_lds(g0, g1, z4, z4, 0);
#endif
}
#endif

// ---------------- weight prep: f32 [3H,K] -> bf16 WMMA-fragment-packed ----------------
// Fragment order: [kb][wave 0..7][which 0..5][lane 0..31][16 cols] (1KB per fragment,
// contiguous per wave). which: 0,1 = r(jt0,1); 2,3 = z; 4,5 = n.
// wcol: [768] f32 = W_ih[:,128] (time channel, applied as rank-1 VALU update).
__global__ void gru_prep_kernel(const float* __restrict__ W_ih,
                                const float* __restrict__ W_hh,
                                unsigned short* __restrict__ WihP,
                                unsigned short* __restrict__ WhhP,
                                float* __restrict__ wcol) {
    const int NIH = 4 * 8 * 6 * 512;   // 98304
    const int NHH = 8 * 8 * 6 * 512;   // 196608
    int i = blockIdx.x * 256 + threadIdx.x;
    if (i < NIH) {
        int c = i & 15, l = (i >> 4) & 31, f = i >> 9;
        int which = f % 6, wvp = (f / 6) & 7, kb = f / 48;
        int n = (which >> 1) * 256 + wvp * 32 + (which & 1) * 16 + (l & 15);
        int k = kb * 32 + ((c >> 3) << 4) + ((l >> 4) << 3) + (c & 7);
        WihP[i] = f2bf(W_ih[n * 129 + k]);
        return;
    }
    int j = i - NIH;
    if (j < NHH) {
        int c = j & 15, l = (j >> 4) & 31, f = j >> 9;
        int which = f % 6, wvp = (f / 6) & 7, kb = f / 48;
        int n = (which >> 1) * 256 + wvp * 32 + (which & 1) * 16 + (l & 15);
        int k = kb * 32 + ((c >> 3) << 4) + ((l >> 4) << 3) + (c & 7);
        WhhP[j] = f2bf(W_hh[n * 256 + k]);
        return;
    }
    int l2 = j - NHH;
    if (l2 < G3) wcol[l2] = W_ih[l2 * 129 + 128];
}

// ---------------- main persistent GRU kernel ----------------
// grid = 16 workgroups; WG owns a 16-row batch tile for all 2048 steps.
// block = 256 threads = 8 wave32's; wave w owns hidden slice [32w, 32w+32)
// (matching r/z/n gate tiles -> whole nonlinearity stays in registers).
__global__ __launch_bounds__(256, 1) void gru_scan_kernel(
    const float* __restrict__ X, const float* __restrict__ Mk, const float* __restrict__ Tt,
    const float* __restrict__ b_ih, const float* __restrict__ b_hh,
    const float* __restrict__ W_lin, const float* __restrict__ b_lin,
    const unsigned short* __restrict__ WihP, const unsigned short* __restrict__ WhhP,
    const float* __restrict__ wcol, float* __restrict__ out) {

    __shared__ __align__(16) float xrawX[2][16][CIN];        // TDM double buffer (X)
    __shared__ __align__(16) float xrawM[2][16][CIN];        // TDM double buffer (M)
    __shared__ __align__(16) unsigned short xbuf[16][128];   // bf16 [Xi|Mi]
    __shared__ __align__(16) unsigned short hbuf[16][256];   // bf16 h (WMMA A source)
    __shared__ float tvals[16];
    __shared__ float obsf[16];
    __shared__ float red[16][16];
    __shared__ float hf[16][256];

    const int tid = threadIdx.x;
    const int wv  = tid >> 5;
    const int ln  = tid & 31;
    const int lh  = ln >> 4;
    const int n16 = ln & 15;
    const int b0  = blockIdx.x * 16;
    const int row = tid >> 4;
    const int seg = tid & 15;

    for (int i = tid; i < 16 * 256; i += 256) ((unsigned short*)hbuf)[i] = 0;

    // h in registers, matching C/D fragment layout: hp[jt][v] = h[M][j],
    // M = v + lh*8, j = wv*32 + jt*16 + n16
    float hp[2][8];
#pragma unroll
    for (int jt = 0; jt < 2; ++jt)
#pragma unroll
        for (int v = 0; v < 8; ++v) hp[jt][v] = 0.0f;

    float br[2], bz[2], bni[2], bnh[2], wcr[2], wcz[2], wcn[2];
#pragma unroll
    for (int jt = 0; jt < 2; ++jt) {
        int j = wv * 32 + jt * 16 + n16;
        br[jt]  = b_ih[j] + b_hh[j];
        bz[jt]  = b_ih[256 + j] + b_hh[256 + j];
        bni[jt] = b_ih[512 + j];
        bnh[jt] = b_hh[512 + j];
        wcr[jt] = wcol[j];
        wcz[jt] = wcol[256 + j];
        wcn[jt] = wcol[512 + j];
    }

    const size_t rowbase = (size_t)(b0 + row) * T_STEPS;

#if USE_TDM
    // prologue: async-stage step 0 into buffer 0 via the Tensor Data Mover
    if (wv == 0) {
        tdm_load_2d_f32(X  + ((size_t)b0 * T_STEPS) * CIN,
                        (unsigned)(unsigned long long)&xrawX[0][0][0], 16, CIN, T_STEPS * CIN);
        tdm_load_2d_f32(Mk + ((size_t)b0 * T_STEPS) * CIN,
                        (unsigned)(unsigned long long)&xrawM[0][0][0], 16, CIN, T_STEPS * CIN);
    }
#endif

    for (int t = 0; t < T_STEPS; ++t) {
        // Launder a zero *offset* (not the pointers!) each iteration: defeats LICM
        // hoist+spill of the 72 weight fragments while keeping the kernarg pointer
        // provenance, so the loads stay global_load_b128 (LOADcnt only), not FLAT.
        size_t lz = 0;
        asm volatile("" : "+s"(lz));
        const unsigned short* wih_t = WihP + lz;
        const unsigned short* whh_t = WhhP + lz;

#if USE_TDM
        const int buf = t & 1;
        if (wv == 0) {
            if (t + 1 < T_STEPS) {
                tdm_load_2d_f32(X  + ((size_t)b0 * T_STEPS + (t + 1)) * CIN,
                                (unsigned)(unsigned long long)&xrawX[buf ^ 1][0][0],
                                16, CIN, T_STEPS * CIN);
                tdm_load_2d_f32(Mk + ((size_t)b0 * T_STEPS + (t + 1)) * CIN,
                                (unsigned)(unsigned long long)&xrawM[buf ^ 1][0][0],
                                16, CIN, T_STEPS * CIN);
                __builtin_amdgcn_s_wait_tensorcnt(2);   // step-t tiles complete
            } else {
                __builtin_amdgcn_s_wait_tensorcnt(0);
            }
        }
        __syncthreads();   // A: xraw[buf] visible to all waves
        {   // ---- LDS f32 -> bf16 convert; obs partial max; tc = t[b]-t[0] ----
            const float4* srcq = (seg < 8) ? (const float4*)&xrawX[buf][row][seg * 8]
                                           : (const float4*)&xrawM[buf][row][(seg - 8) * 8];
            float4 a0 = srcq[0];
            float4 a1 = srcq[1];
#else
        {   // ---- direct global staging fallback ----
            const float* src = (seg < 8) ? (X  + (rowbase + t) * CIN + seg * 8)
                                         : (Mk + (rowbase + t) * CIN + (seg - 8) * 8);
            float4 a0 = ((const float4*)src)[0];
            float4 a1 = ((const float4*)src)[1];
#endif
            uint4 u;
            u.x = f2bf(a0.x) | ((unsigned)f2bf(a0.y) << 16);
            u.y = f2bf(a0.z) | ((unsigned)f2bf(a0.w) << 16);
            u.z = f2bf(a1.x) | ((unsigned)f2bf(a1.y) << 16);
            u.w = f2bf(a1.z) | ((unsigned)f2bf(a1.w) << 16);
            *(uint4*)&xbuf[row][seg * 8] = u;
            float pm = -1.0f;
            if (seg >= 8)
                pm = fmaxf(fmaxf(fmaxf(a0.x, a0.y), fmaxf(a0.z, a0.w)),
                           fmaxf(fmaxf(a1.x, a1.y), fmaxf(a1.z, a1.w)));
            red[row][seg] = pm;
            if (tid < 16) tvals[tid] = Tt[(size_t)(b0 + tid) * T_STEPS + t] - Tt[t];
        }
        __syncthreads();   // B: xbuf/red/tvals ready; prev hbuf stores visible
        if (tid < 16) {
            float mx = red[tid][8];
#pragma unroll
            for (int k = 9; k < 16; ++k) mx = fmaxf(mx, red[tid][k]);
            obsf[tid] = (mx > 0.5f) ? 1.0f : 0.0f;
        }

        v8f accr[2], accz[2], accni[2], accnh[2];
#pragma unroll
        for (int jt = 0; jt < 2; ++jt) {
            accr[jt]  = (v8f){0, 0, 0, 0, 0, 0, 0, 0};
            accz[jt]  = (v8f){0, 0, 0, 0, 0, 0, 0, 0};
            accni[jt] = (v8f){0, 0, 0, 0, 0, 0, 0, 0};
            accnh[jt] = (v8f){0, 0, 0, 0, 0, 0, 0, 0};
        }

        // ---- gi: x[16,128] @ Wih slice, 4 K-blocks, packed coalesced fragments ----
#pragma unroll
        for (int kb = 0; kb < 4; ++kb) {
            Frag fa;
            const int c0 = kb * 32 + lh * 8;
            fa.q[0] = *(const uint4*)&xbuf[n16][c0];
            fa.q[1] = *(const uint4*)&xbuf[n16][c0 + 16];
            const unsigned short* bp = wih_t + ((size_t)(kb * 8 + wv) * 6) * 512 + ln * 16;
            Frag f0, f1, f2, f3, f4, f5;
            f0.q[0] = *(const uint4*)(bp + 0 * 512);  f0.q[1] = *(const uint4*)(bp + 0 * 512 + 8);
            f1.q[0] = *(const uint4*)(bp + 1 * 512);  f1.q[1] = *(const uint4*)(bp + 1 * 512 + 8);
            f2.q[0] = *(const uint4*)(bp + 2 * 512);  f2.q[1] = *(const uint4*)(bp + 2 * 512 + 8);
            f3.q[0] = *(const uint4*)(bp + 3 * 512);  f3.q[1] = *(const uint4*)(bp + 3 * 512 + 8);
            f4.q[0] = *(const uint4*)(bp + 4 * 512);  f4.q[1] = *(const uint4*)(bp + 4 * 512 + 8);
            f5.q[0] = *(const uint4*)(bp + 5 * 512);  f5.q[1] = *(const uint4*)(bp + 5 * 512 + 8);
            accr[0]  = __builtin_amdgcn_wmma_f32_16x16x32_bf16(false, fa.v, false, f0.v, (short)0, accr[0],  false, false);
            accr[1]  = __builtin_amdgcn_wmma_f32_16x16x32_bf16(false, fa.v, false, f1.v, (short)0, accr[1],  false, false);
            accz[0]  = __builtin_amdgcn_wmma_f32_16x16x32_bf16(false, fa.v, false, f2.v, (short)0, accz[0],  false, false);
            accz[1]  = __builtin_amdgcn_wmma_f32_16x16x32_bf16(false, fa.v, false, f3.v, (short)0, accz[1],  false, false);
            accni[0] = __builtin_amdgcn_wmma_f32_16x16x32_bf16(false, fa.v, false, f4.v, (short)0, accni[0], false, false);
            accni[1] = __builtin_amdgcn_wmma_f32_16x16x32_bf16(false, fa.v, false, f5.v, (short)0, accni[1], false, false);
        }

        // ---- gh: h[16,256] @ Whh slice, 8 K-blocks ----
#pragma unroll
        for (int kb = 0; kb < 8; ++kb) {
            Frag fa;
            const int c0 = kb * 32 + lh * 8;
            fa.q[0] = *(const uint4*)&hbuf[n16][c0];
            fa.q[1] = *(const uint4*)&hbuf[n16][c0 + 16];
            const unsigned short* bp = whh_t + ((size_t)(kb * 8 + wv) * 6) * 512 + ln * 16;
            Frag f0, f1, f2, f3, f4, f5;
            f0.q[0] = *(const uint4*)(bp + 0 * 512);  f0.q[1] = *(const uint4*)(bp + 0 * 512 + 8);
            f1.q[0] = *(const uint4*)(bp + 1 * 512);  f1.q[1] = *(const uint4*)(bp + 1 * 512 + 8);
            f2.q[0] = *(const uint4*)(bp + 2 * 512);  f2.q[1] = *(const uint4*)(bp + 2 * 512 + 8);
            f3.q[0] = *(const uint4*)(bp + 3 * 512);  f3.q[1] = *(const uint4*)(bp + 3 * 512 + 8);
            f4.q[0] = *(const uint4*)(bp + 4 * 512);  f4.q[1] = *(const uint4*)(bp + 4 * 512 + 8);
            f5.q[0] = *(const uint4*)(bp + 5 * 512);  f5.q[1] = *(const uint4*)(bp + 5 * 512 + 8);
            accr[0]  = __builtin_amdgcn_wmma_f32_16x16x32_bf16(false, fa.v, false, f0.v, (short)0, accr[0],  false, false);
            accr[1]  = __builtin_amdgcn_wmma_f32_16x16x32_bf16(false, fa.v, false, f1.v, (short)0, accr[1],  false, false);
            accz[0]  = __builtin_amdgcn_wmma_f32_16x16x32_bf16(false, fa.v, false, f2.v, (short)0, accz[0],  false, false);
            accz[1]  = __builtin_amdgcn_wmma_f32_16x16x32_bf16(false, fa.v, false, f3.v, (short)0, accz[1],  false, false);
            accnh[0] = __builtin_amdgcn_wmma_f32_16x16x32_bf16(false, fa.v, false, f4.v, (short)0, accnh[0], false, false);
            accnh[1] = __builtin_amdgcn_wmma_f32_16x16x32_bf16(false, fa.v, false, f5.v, (short)0, accnh[1], false, false);
        }

        // ---- gates + candidate h, in registers ----
        float tv[8];
#pragma unroll
        for (int v = 0; v < 8; ++v) tv[v] = tvals[v + lh * 8];
        float hnew[2][8];
#pragma unroll
        for (int jt = 0; jt < 2; ++jt)
#pragma unroll
            for (int v = 0; v < 8; ++v) {
                float r   = sigmoidf_(accr[jt][v] + tv[v] * wcr[jt] + br[jt]);
                float z   = sigmoidf_(accz[jt][v] + tv[v] * wcz[jt] + bz[jt]);
                float xin = accni[jt][v] + tv[v] * wcn[jt] + bni[jt];
                float hn  = accnh[jt][v] + bnh[jt];
                float nn  = tanhf_(xin + r * hn);
                hnew[jt][v] = (1.0f - z) * nn + z * hp[jt][v];
            }
        __syncthreads();   // C: all waves done reading hbuf
#pragma unroll
        for (int jt = 0; jt < 2; ++jt)
#pragma unroll
            for (int v = 0; v < 8; ++v) {
                float o = obsf[v + lh * 8];
                hp[jt][v] = o * hnew[jt][v] + (1.0f - o) * hp[jt][v];
                hbuf[v + lh * 8][wv * 32 + jt * 16 + n16] = f2bf(hp[jt][v]);
            }
    }

    // ---- out = h @ W_lin.T + b_lin (tiny: 16 rows x 2 outputs) ----
#pragma unroll
    for (int jt = 0; jt < 2; ++jt)
#pragma unroll
        for (int v = 0; v < 8; ++v)
            hf[v + lh * 8][wv * 32 + jt * 16 + n16] = hp[jt][v];
    __syncthreads();
    if (tid < 32) {
        int m = tid >> 1, o = tid & 1;
        float s = b_lin[o];
        for (int j = 0; j < HID; ++j) s += hf[m][j] * W_lin[o * HID + j];
        out[(size_t)(b0 + m) * 2 + o] = s;
    }
}

extern "C" void kernel_launch(void* const* d_in, const int* in_sizes, int n_in,
                              void* d_out, int out_size, void* d_ws, size_t ws_size,
                              hipStream_t stream) {
    const float* X     = (const float*)d_in[0];
    const float* Mk    = (const float*)d_in[1];
    const float* Tt    = (const float*)d_in[2];
    const float* W_ih  = (const float*)d_in[3];
    const float* W_hh  = (const float*)d_in[4];
    const float* b_ih  = (const float*)d_in[5];
    const float* b_hh  = (const float*)d_in[6];
    const float* W_lin = (const float*)d_in[7];
    const float* b_lin = (const float*)d_in[8];
    float* out = (float*)d_out;

    char* ws = (char*)d_ws;
    unsigned short* WihP = (unsigned short*)ws;                // 196608 B
    unsigned short* WhhP = (unsigned short*)(ws + 196608);     // 393216 B
    float*          wcol = (float*)(ws + 196608 + 393216);     // 3072 B

    const int total = 4 * 8 * 6 * 512 + 8 * 8 * 6 * 512 + G3; // 295680
    gru_prep_kernel<<<(total + 255) / 256, 256, 0, stream>>>(W_ih, W_hh, WihP, WhhP, wcol);
    gru_scan_kernel<<<16, 256, 0, stream>>>(X, Mk, Tt, b_ih, b_hh, W_lin, b_lin,
                                            WihP, WhhP, wcol, out);
}
